// PointTransformerLayer_15539191676961
// MI455X (gfx1250) — matrix operand order repaired and checked by reading, not compiled, not run on hardware
//
#include <hip/hip_runtime.h>

#define NPTS 16384
#define KNBR 16
#define CH   64
#define NK   (NPTS * KNBR)
#define BNEPS 1e-5f

#define CHUNK 2048                 // points staged per LDS buffer
#define NCHUNK (NPTS / CHUNK)      // 8

typedef __attribute__((ext_vector_type(16))) __bf16 v16bf;
typedef __attribute__((ext_vector_type(8)))  float  v8f;
typedef __attribute__((ext_vector_type(2)))  float  v2f;

__device__ __forceinline__ __bf16 f2bf(float x) { return (__bf16)x; }

__device__ __forceinline__ v8f wmma_bf16(v16bf a, v16bf b, v8f c) {
  // D = A(16x32 bf16) * B(32x16 bf16) + C(f32)
  return __builtin_amdgcn_wmma_f32_16x16x32_bf16(false, a, false, b, (short)0, c,
                                                 false, false);
}

// B-matrix tile (32x16, K-major) from a row-major (64,64) fp32 weight.
// Lane layout: col = lane&15, khalf = lane>>4; elem e<8 -> K 8*khalf+e, e>=8 -> K 16+8*khalf+(e-8).
__device__ __forceinline__ v16bf load_bmat(const float* __restrict__ W, int k0,
                                           int col, int khalf) {
  v16bf b;
#pragma unroll
  for (int e = 0; e < 16; ++e)
    b[e] = f2bf(W[(k0 + (e < 8 ? khalf * 8 + e : 16 + khalf * 8 + (e - 8))) * CH + col]);
  return b;
}

// Async global->LDS copy of one 24KB point chunk (CDNA5 ASYNCcnt path).
// 1536 x B128 transfers spread over 256 threads (6 each).
__device__ __forceinline__ void async_fetch_chunk(const float* lbuf,
                                                  const float* gsrc, int t) {
  // low 32 bits of a generic LDS pointer are the LDS byte offset
  unsigned lds = (unsigned)(unsigned long long)lbuf + (unsigned)t * 16u;
  const char* g = (const char*)gsrc + (size_t)t * 16;
#pragma unroll
  for (int r = 0; r < 6; ++r) {
    asm volatile("global_load_async_to_lds_b128 %0, %1, off"
                 :: "v"(lds), "v"(g)
                 : "memory");
    lds += 4096;
    g += 4096;
  }
}

__device__ __forceinline__ void wait_async0() {
  asm volatile("s_wait_asynccnt 0x0" ::: "memory");
}

// ---------------------------------------------------------------------------
// 1) QKV GEMM: (N x 64) * (64 x 192) -> qkv (N x 192), bf16 WMMA, fp32 accum
// ---------------------------------------------------------------------------
__global__ __launch_bounds__(256) void qkv_gemm_kernel(
    const float* __restrict__ feat, const float* __restrict__ Wq,
    const float* __restrict__ Wk, const float* __restrict__ Wv,
    float* __restrict__ qkv) {
  const int lane = threadIdx.x & 31, half = lane >> 4, l15 = lane & 15;
  const int wave = (int)((blockIdx.x * blockDim.x + threadIdx.x) >> 5);
  const int mt = wave / 12, nt = wave % 12;
  if (mt >= NPTS / 16) return;
  const float* W = (nt < 4) ? Wq : (nt < 8) ? Wk : Wv;
  const int ncol = (nt & 3) * 16 + l15;
  const float* arow = feat + (size_t)(mt * 16 + l15) * CH;
  v16bf a0, a1;
#pragma unroll
  for (int e = 0; e < 8; ++e) {
    a0[e]     = f2bf(arow[half * 8 + e]);
    a0[e + 8] = f2bf(arow[16 + half * 8 + e]);
    a1[e]     = f2bf(arow[32 + half * 8 + e]);
    a1[e + 8] = f2bf(arow[48 + half * 8 + e]);
  }
  v16bf b0 = load_bmat(W, 0, ncol, half);
  v16bf b1 = load_bmat(W, 32, ncol, half);
  v8f c = {};
  c = wmma_bf16(a0, b0, c);
  c = wmma_bf16(a1, b1, c);
#pragma unroll
  for (int r = 0; r < 8; ++r)
    qkv[(size_t)(mt * 16 + half * 8 + r) * 192 + nt * 16 + l15] = c[r];
}

// ---------------------------------------------------------------------------
// 2) KNN: one wave per query. Points are double-buffered through LDS with
//    GLOBAL_LOAD_ASYNC_TO_LDS_B128 (scan buf[c] while buf[c^1] fills).
//    Per-lane sorted top-16 over strided candidates, then 16 rounds of
//    tie-broken wave argmin to merge.
// ---------------------------------------------------------------------------
__global__ __launch_bounds__(256) void knn_kernel(const float* __restrict__ pts,
                                                  int* __restrict__ knn) {
  __shared__ __align__(16) float spts[2][CHUNK * 3];  // 2 x 24KB
  const int t = threadIdx.x;
  const int lane = t & 31;
  const int i = (int)blockIdx.x * 8 + (t >> 5);  // grid is exact: i < NPTS
  const float qx = pts[i * 3 + 0], qy = pts[i * 3 + 1], qz = pts[i * 3 + 2];

  float bd[16];
  int bi[16];
#pragma unroll
  for (int s = 0; s < 16; ++s) { bd[s] = 3.0e38f; bi[s] = 0; }

  async_fetch_chunk(&spts[0][0], pts, t);  // chunk 0 in flight

  for (int c = 0; c < NCHUNK; ++c) {
    wait_async0();      // my piece of chunk c has landed
    __syncthreads();    // everyone's piece has landed; buf[(c+1)&1] is idle
    if (c + 1 < NCHUNK)
      async_fetch_chunk(&spts[(c + 1) & 1][0], pts + (size_t)(c + 1) * CHUNK * 3, t);

    const float* lp = &spts[c & 1][0];
    const int cbase = c * CHUNK;
    for (int l = lane; l < CHUNK; l += 32) {
      const float dx = qx - lp[l * 3 + 0];
      const float dy = qy - lp[l * 3 + 1];
      const float dz = qz - lp[l * 3 + 2];
      const float d2 = dx * dx + dy * dy + dz * dz;
      if (d2 < bd[15]) {  // bubble-insert into per-lane sorted list
        float cv = d2;
        int ci = cbase + l;
#pragma unroll
        for (int s = 0; s < 16; ++s) {
          if (cv < bd[s]) {
            float td = bd[s]; int ti = bi[s];
            bd[s] = cv; bi[s] = ci;
            cv = td; ci = ti;
          }
        }
      }
    }
  }

  // merge the 32 sorted lists: 16 rounds of wave argmin over lane heads
  for (int r = 0; r < 16; ++r) {
    float v = bd[0];
    int vi = bi[0];
    int wl = lane;
#pragma unroll
    for (int off = 16; off; off >>= 1) {
      const float ov = __shfl_xor(v, off, 32);
      const int ovi = __shfl_xor(vi, off, 32);
      const int owl = __shfl_xor(wl, off, 32);
      if (ov < v || (ov == v && owl < wl)) { v = ov; vi = ovi; wl = owl; }
    }
    if (lane == 0) knn[i * KNBR + r] = vi;
    if (lane == wl) {  // winner pops its head
#pragma unroll
      for (int s = 0; s < 15; ++s) { bd[s] = bd[s + 1]; bi[s] = bi[s + 1]; }
      bd[15] = 3.0e38f;
    }
  }
}

// ---------------------------------------------------------------------------
// Deterministic BN statistics (fixed 256 blocks -> fixed-order finalize)
// ---------------------------------------------------------------------------
__device__ __forceinline__ float block_reduce_add(float v, float* sh) {
  const int t = threadIdx.x;
  sh[t] = v;
  __syncthreads();
  for (int o = 128; o > 0; o >>= 1) {
    if (t < o) sh[t] += sh[t + o];
    __syncthreads();
  }
  const float r = sh[0];
  __syncthreads();
  return r;
}

__global__ __launch_bounds__(256) void stats_pos_kernel(
    const float* __restrict__ pts, const int* __restrict__ knn,
    const float* __restrict__ pW1, float* __restrict__ part) {
  __shared__ float red[256];
  const int t = threadIdx.x;
  float s[3] = {0.f, 0.f, 0.f}, q[3] = {0.f, 0.f, 0.f};
  for (int r = (int)blockIdx.x * 256 + t; r < NK; r += 256 * 256) {
    const int i = r >> 4;
    const int j = knn[r];
    const float rx = pts[i * 3 + 0] - pts[j * 3 + 0];
    const float ry = pts[i * 3 + 1] - pts[j * 3 + 1];
    const float rz = pts[i * 3 + 2] - pts[j * 3 + 2];
#pragma unroll
    for (int c = 0; c < 3; ++c) {
      const float h = rx * pW1[c] + ry * pW1[3 + c] + rz * pW1[6 + c];
      s[c] += h;
      q[c] += h * h;
    }
  }
#pragma unroll
  for (int c = 0; c < 3; ++c) {
    const float rs = block_reduce_add(s[c], red);
    const float rq = block_reduce_add(q[c], red);
    if (t == 0) {
      part[blockIdx.x * 6 + c] = rs;
      part[blockIdx.x * 6 + 3 + c] = rq;
    }
  }
}

// mode 0: vec_sim recomputed on the fly; mode 1: read src directly
__global__ __launch_bounds__(256) void stats64_kernel(
    int mode, const float* __restrict__ qkv, const int* __restrict__ knn,
    const float* __restrict__ rpe, const float* __restrict__ src,
    float* __restrict__ part) {
  __shared__ float sh[8][64];
  const int t = threadIdx.x;
  const int c = t & 63;
  const int g = t >> 6;
  float s = 0.f, q = 0.f;
  for (int r = (int)blockIdx.x * 4 + g; r < NK; r += 1024) {
    float v;
    if (mode == 0) {
      const int i = r >> 4;
      const int j = knn[r];
      v = qkv[(size_t)i * 192 + c] - qkv[(size_t)j * 192 + 64 + c] +
          rpe[(size_t)r * CH + c];
    } else {
      v = src[(size_t)r * CH + c];
    }
    s += v;
    q += v * v;
  }
  sh[g][c] = s;
  sh[4 + g][c] = q;
  __syncthreads();
  if (t < 64) {
    const float ss = sh[0][t] + sh[1][t] + sh[2][t] + sh[3][t];
    const float qq = sh[4][t] + sh[5][t] + sh[6][t] + sh[7][t];
    part[blockIdx.x * 128 + t] = ss;
    part[blockIdx.x * 128 + 64 + t] = qq;
  }
}

__global__ void finalize_bn_kernel(const float* __restrict__ part, int nb, int ch,
                                   float invM, const float* __restrict__ g,
                                   const float* __restrict__ b,
                                   float* __restrict__ scale,
                                   float* __restrict__ shift) {
  const int c = threadIdx.x;
  if (c >= ch) return;
  float s = 0.f, q = 0.f;
  for (int blk = 0; blk < nb; ++blk) {
    s += part[blk * 2 * ch + c];
    q += part[blk * 2 * ch + ch + c];
  }
  const float mu = s * invM;
  const float var = q * invM - mu * mu;
  const float sc = g[c] * rsqrtf(var + BNEPS);
  scale[c] = sc;
  shift[c] = b[c] - mu * sc;
}

// ---------------------------------------------------------------------------
// 4) rel-pos encode: relu(bn(rel@pW1)) @ pW2  via V_WMMA_F32_16X16X4_F32 (K=3+pad)
// ---------------------------------------------------------------------------
__global__ __launch_bounds__(256) void rpe_gemm_kernel(
    const float* __restrict__ pts, const int* __restrict__ knn,
    const float* __restrict__ pW1, const float* __restrict__ pW2,
    const float* __restrict__ ps, const float* __restrict__ pt,
    float* __restrict__ rpe) {
  const int lane = threadIdx.x & 31, half = lane >> 4, l15 = lane & 15;
  const int wave = (int)((blockIdx.x * blockDim.x + threadIdx.x) >> 5);
  const int mt = wave >> 2, nt = wave & 3;
  if (mt >= NK / 16) return;
  const int row = mt * 16 + l15;
  const int i = row >> 4;
  const int j = knn[row];
  const float rx = pts[i * 3 + 0] - pts[j * 3 + 0];
  const float ry = pts[i * 3 + 1] - pts[j * 3 + 1];
  const float rz = pts[i * 3 + 2] - pts[j * 3 + 2];
  float hb[4];
#pragma unroll
  for (int cc = 0; cc < 3; ++cc) {
    const float h = rx * pW1[cc] + ry * pW1[3 + cc] + rz * pW1[6 + cc];
    hb[cc] = fmaxf(h * ps[cc] + pt[cc], 0.f);
  }
  hb[3] = 0.f;  // K=3 padded to 4
  const int n = nt * 16 + l15;
  v2f a, b;
  a[0] = hb[half * 2 + 0];
  a[1] = hb[half * 2 + 1];
  b[0] = pW2[(half * 2 + 0) * CH + n];
  b[1] = (half * 2 + 1 < 3) ? pW2[(half * 2 + 1) * CH + n] : 0.f;
  v8f c = {};
  c = __builtin_amdgcn_wmma_f32_16x16x4_f32(false, a, false, b, (short)0, c,
                                            false, false);
#pragma unroll
  for (int r = 0; r < 8; ++r)
    rpe[(size_t)(mt * 16 + half * 8 + r) * CH + n] = c[r];
}

// ---------------------------------------------------------------------------
// 6) attn layer 1: relu(bn1(vsim)) @ a_W1  (vsim gathered on the fly), bf16 WMMA
// ---------------------------------------------------------------------------
__global__ __launch_bounds__(256) void attn1_gemm_kernel(
    const float* __restrict__ qkv, const int* __restrict__ knn,
    const float* __restrict__ rpe, const float* __restrict__ W1,
    const float* __restrict__ s1, const float* __restrict__ t1,
    float* __restrict__ h1) {
  const int lane = threadIdx.x & 31, half = lane >> 4, l15 = lane & 15;
  const int wave = (int)((blockIdx.x * blockDim.x + threadIdx.x) >> 5);
  const int mt = wave >> 2, nt = wave & 3;
  if (mt >= NK / 16) return;
  const int row = mt * 16 + l15;
  const int i = row >> 4;
  const int j = knn[row];
  const float* fq = qkv + (size_t)i * 192;
  const float* fk = qkv + (size_t)j * 192 + 64;
  const float* rp = rpe + (size_t)row * CH;
  v16bf a0, a1;
#pragma unroll
  for (int e = 0; e < 8; ++e) {
    const int ka = half * 8 + e;
    const int kb = 16 + half * 8 + e;
    const int kc = 32 + half * 8 + e;
    const int kd = 48 + half * 8 + e;
    a0[e]     = f2bf(fmaxf((fq[ka] - fk[ka] + rp[ka]) * s1[ka] + t1[ka], 0.f));
    a0[e + 8] = f2bf(fmaxf((fq[kb] - fk[kb] + rp[kb]) * s1[kb] + t1[kb], 0.f));
    a1[e]     = f2bf(fmaxf((fq[kc] - fk[kc] + rp[kc]) * s1[kc] + t1[kc], 0.f));
    a1[e + 8] = f2bf(fmaxf((fq[kd] - fk[kd] + rp[kd]) * s1[kd] + t1[kd], 0.f));
  }
  const int n = nt * 16 + l15;
  v16bf b0 = load_bmat(W1, 0, n, half);
  v16bf b1 = load_bmat(W1, 32, n, half);
  v8f c = {};
  c = wmma_bf16(a0, b0, c);
  c = wmma_bf16(a1, b1, c);
#pragma unroll
  for (int r = 0; r < 8; ++r)
    h1[(size_t)(mt * 16 + half * 8 + r) * CH + n] = c[r];
}

// ---------------------------------------------------------------------------
// 8) attn layer 2 in-place: logits = relu(bn2(h1)) @ a_W2 + a_B2 (bf16 WMMA).
//    Wave reads its 16 rows fully into registers before writing -> safe.
// ---------------------------------------------------------------------------
__global__ __launch_bounds__(256) void attn2_gemm_kernel(
    float* __restrict__ bufA, const float* __restrict__ W2,
    const float* __restrict__ B2, const float* __restrict__ s2,
    const float* __restrict__ t2) {
  const int lane = threadIdx.x & 31, half = lane >> 4, l15 = lane & 15;
  const int mt = (int)((blockIdx.x * blockDim.x + threadIdx.x) >> 5);
  if (mt >= NK / 16) return;
  const int m0 = mt * 16;
  const float* hrow = bufA + (size_t)(m0 + l15) * CH;
  v16bf a0, a1;
#pragma unroll
  for (int e = 0; e < 8; ++e) {
    const int ka = half * 8 + e;
    const int kb = 16 + half * 8 + e;
    const int kc = 32 + half * 8 + e;
    const int kd = 48 + half * 8 + e;
    a0[e]     = f2bf(fmaxf(hrow[ka] * s2[ka] + t2[ka], 0.f));
    a0[e + 8] = f2bf(fmaxf(hrow[kb] * s2[kb] + t2[kb], 0.f));
    a1[e]     = f2bf(fmaxf(hrow[kc] * s2[kc] + t2[kc], 0.f));
    a1[e + 8] = f2bf(fmaxf(hrow[kd] * s2[kd] + t2[kd], 0.f));
  }
  v8f cs[4];
#pragma unroll
  for (int nt = 0; nt < 4; ++nt) {
    v16bf b0 = load_bmat(W2, 0, nt * 16 + l15, half);
    v16bf b1 = load_bmat(W2, 32, nt * 16 + l15, half);
    v8f c = {};
    c = wmma_bf16(a0, b0, c);
    c = wmma_bf16(a1, b1, c);
    cs[nt] = c;
  }
#pragma unroll
  for (int nt = 0; nt < 4; ++nt) {
    const float bias = B2[nt * 16 + l15];
#pragma unroll
    for (int r = 0; r < 8; ++r)
      bufA[(size_t)(m0 + half * 8 + r) * CH + nt * 16 + l15] = cs[nt][r] + bias;
  }
}

// ---------------------------------------------------------------------------
// 9) softmax over k=16 per (query, channel) + weighted gather-sum -> out
// ---------------------------------------------------------------------------
__global__ __launch_bounds__(256) void softmax_out_kernel(
    const float* __restrict__ logits, const float* __restrict__ qkv,
    const int* __restrict__ knn, const float* __restrict__ rpe,
    float* __restrict__ out) {
  const int lane = threadIdx.x & 31;
  const int i = (int)((blockIdx.x * blockDim.x + threadIdx.x) >> 5);
  if (i >= NPTS) return;
#pragma unroll
  for (int cc = 0; cc < 2; ++cc) {
    const int c = cc * 32 + lane;
    float lv[16];
    float m = -3.0e38f;
#pragma unroll
    for (int kk = 0; kk < 16; ++kk) {
      lv[kk] = logits[(size_t)(i * KNBR + kk) * CH + c];
      m = fmaxf(m, lv[kk]);
    }
    float s = 0.f;
#pragma unroll
    for (int kk = 0; kk < 16; ++kk) {
      lv[kk] = __expf(lv[kk] - m);
      s += lv[kk];
    }
    const float inv = 1.f / s;
    float acc = 0.f;
#pragma unroll
    for (int kk = 0; kk < 16; ++kk) {
      const int j = knn[i * KNBR + kk];
      acc += lv[kk] * (qkv[(size_t)j * 192 + 128 + c] +
                       rpe[(size_t)(i * KNBR + kk) * CH + c]);
    }
    out[(size_t)i * CH + c] = acc * inv;
  }
}

// ---------------------------------------------------------------------------
extern "C" void kernel_launch(void* const* d_in, const int* in_sizes, int n_in,
                              void* d_out, int out_size, void* d_ws, size_t ws_size,
                              hipStream_t stream) {
  const float* points = (const float*)d_in[0];
  const float* feats  = (const float*)d_in[1];
  const float* Wq = (const float*)d_in[2];
  const float* Wk = (const float*)d_in[3];
  const float* Wv = (const float*)d_in[4];
  const float* a_g1 = (const float*)d_in[5];
  const float* a_b1 = (const float*)d_in[6];
  const float* a_W1 = (const float*)d_in[7];
  const float* a_g2 = (const float*)d_in[8];
  const float* a_b2 = (const float*)d_in[9];
  const float* a_W2 = (const float*)d_in[10];
  const float* a_B2 = (const float*)d_in[11];
  const float* p_W1 = (const float*)d_in[12];
  const float* p_g  = (const float*)d_in[13];
  const float* p_b  = (const float*)d_in[14];
  const float* p_W2 = (const float*)d_in[15];
  float* out = (float*)d_out;

  // workspace layout (floats)
  float* ws   = (float*)d_ws;
  float* qkv  = ws;                                   // N*192
  int*   knn  = (int*)(ws + (size_t)NPTS * 192);      // N*K
  float* rpe  = ws + (size_t)NPTS * 192 + NK;         // NK*64
  float* bufA = rpe + (size_t)NK * CH;                // NK*64 (h1, then logits)
  float* part = bufA + (size_t)NK * CH;               // 256*128 partials
  float* bn1s = part + 256 * 128;
  float* bn1t = bn1s + CH;
  float* bn2s = bn1t + CH;
  float* bn2t = bn2s + CH;
  float* ps   = bn2t + CH;  // 4
  float* pt   = ps + 4;     // 4

  const float invM = 1.0f / (float)NK;

  qkv_gemm_kernel<<<1536, 256, 0, stream>>>(feats, Wq, Wk, Wv, qkv);
  knn_kernel<<<2048, 256, 0, stream>>>(points, knn);
  stats_pos_kernel<<<256, 256, 0, stream>>>(points, knn, p_W1, part);
  finalize_bn_kernel<<<1, 64, 0, stream>>>(part, 256, 3, invM, p_g, p_b, ps, pt);
  rpe_gemm_kernel<<<8192, 256, 0, stream>>>(points, knn, p_W1, p_W2, ps, pt, rpe);
  stats64_kernel<<<256, 256, 0, stream>>>(0, qkv, knn, rpe, bufA, part);
  finalize_bn_kernel<<<1, 64, 0, stream>>>(part, 256, 64, invM, a_g1, a_b1, bn1s, bn1t);
  attn1_gemm_kernel<<<8192, 256, 0, stream>>>(qkv, knn, rpe, a_W1, bn1s, bn1t, bufA);
  stats64_kernel<<<256, 256, 0, stream>>>(1, qkv, knn, rpe, bufA, part);
  finalize_bn_kernel<<<1, 64, 0, stream>>>(part, 256, 64, invM, a_g2, a_b2, bn2s, bn2t);
  attn2_gemm_kernel<<<2048, 256, 0, stream>>>(bufA, a_W2, a_B2, bn2s, bn2t);
  softmax_out_kernel<<<2048, 256, 0, stream>>>(bufA, qkv, knn, rpe, out);

  (void)in_sizes; (void)n_in; (void)out_size; (void)ws_size;
}
